// MultiScaleTemporalAttention_52063593562207
// MI455X (gfx1250) — matrix operand order repaired and checked by reading, not compiled, not run on hardware
//
#include <hip/hip_runtime.h>
#include <hip/hip_bf16.h>

// ---------------------------------------------------------------------------
// MultiScaleTemporalAttention for MI455X (gfx1250), bf16 WMMA everywhere.
// B=2 S=2048 D=1024 H=16 HD=64 WINDOW=64 DECAY=0.9
// ---------------------------------------------------------------------------

typedef __bf16 bf16_t;
typedef __attribute__((ext_vector_type(16))) __bf16 v16bf;
typedef __attribute__((ext_vector_type(8)))  __bf16 v8bf;
typedef __attribute__((ext_vector_type(8)))  float  v8f;

#define Bn   2
#define Sn   2048
#define Dn   1024
#define Hn   16
#define HDn  64
#define Mrows (Bn * Sn)                 // 4096
#define SZE  ((size_t)Mrows * Dn)       // elements per QKV buffer

#define SCALEF  0.125f                  // 1/sqrt(64)
#define LOGD    (-0.10536051565782628f) // log(0.9)
#define NEGB    (-1e30f)

__device__ __forceinline__ bf16_t f2bf(float f) {
  unsigned int x = __float_as_uint(f);
  x = x + 0x7FFFu + ((x >> 16) & 1u);          // round-to-nearest-even
  unsigned short s = (unsigned short)(x >> 16);
  bf16_t b;
  __builtin_memcpy(&b, &s, sizeof(b));
  return b;
}

__device__ __forceinline__ v8f vzero8() {
  v8f v;
#pragma unroll
  for (int i = 0; i < 8; ++i) v[i] = 0.0f;
  return v;
}

// D = A*B + C, bf16 inputs, f32 accum
__device__ __forceinline__ v8f wmma_bf(v16bf a, v16bf b, v8f c) {
  return __builtin_amdgcn_wmma_f32_16x16x32_bf16(false, a, false, b,
                                                 (short)0, c, false, false);
}

// A-operand (16x32, M=lane&15): per-lane elements are two contiguous 16B runs:
//   a[e] = A[m][kbase + half*16 + e]
__device__ __forceinline__ v16bf load_a16x32(const bf16_t* __restrict__ row,
                                             int kbase, int half) {
  v8bf lo = *(const v8bf*)(row + kbase + half * 16);
  v8bf hi = *(const v8bf*)(row + kbase + half * 16 + 8);
  v16bf a;
#pragma unroll
  for (int i = 0; i < 8; ++i) { a[i] = lo[i]; a[i + 8] = hi[i]; }
  return a;
}

// ---------------------------------------------------------------------------
// Kernel 1: fp32 -> bf16 elementwise convert (for x)
// ---------------------------------------------------------------------------
__global__ void cvt_f32_bf16(const float* __restrict__ src,
                             bf16_t* __restrict__ dst, int n) {
  int i = blockIdx.x * blockDim.x + threadIdx.x;
  if (i < n) dst[i] = f2bf(src[i]);
}

// ---------------------------------------------------------------------------
// Kernel 2: pack a (K x 1024) fp32 weight into WMMA B-operand layout:
//   Wp[((tn*KT + tk)*32 + lane)*16 + e] = bf16( W[(tk*32 + half*16 + e)*1024 + tn*16 + (lane&15)] )
// so the GEMM loads B as one contiguous v16bf per lane.
// ---------------------------------------------------------------------------
__global__ void pack_w_b(const float* __restrict__ W,
                         bf16_t* __restrict__ Wp, int K) {
  const int KT = K / 32;
  const int total = (Dn / 16) * KT * 32 * 16;
  int idx = blockIdx.x * blockDim.x + threadIdx.x;
  if (idx >= total) return;
  int e    = idx & 15;
  int lane = (idx >> 4) & 31;
  int rest = idx >> 9;
  int tk   = rest % KT;
  int tn   = rest / KT;
  int half = lane >> 4;
  int n    = tn * 16 + (lane & 15);
  int k    = tk * 32 + half * 16 + e;
  Wp[idx] = f2bf(W[(size_t)k * Dn + n]);
}

// ---------------------------------------------------------------------------
// Software-pipelined 16(M) x 64(N) GEMM strip body (shared by kernels 3 & 5).
// Loads for k-step t+1 are issued before the WMMAs of step t, so VMEM flies
// under the matrix pipe instead of draining before every WMMA.
// KT must be a power of two.
// ---------------------------------------------------------------------------
__device__ __forceinline__ void gemm_strip_16x64(
    const bf16_t* __restrict__ arow,   // A row base (row m = lane&15)
    const bf16_t* __restrict__ Wbase,  // packed B + lane*16
    int KT, int half, v8f acc[4]) {
  v16bf a_cur = load_a16x32(arow, 0, half);
  v16bf b_cur[4];
#pragma unroll
  for (int j = 0; j < 4; ++j)
    b_cur[j] = *(const v16bf*)(Wbase + ((size_t)j * KT) * 512);

#pragma unroll 2
  for (int tk = 0; tk < KT; ++tk) {
    const int tkn = (tk + 1) & (KT - 1);   // last iter reloads tile 0 (benign)
    __builtin_prefetch(arow + tk * 32 + 256, 0, 3);
    v16bf a_nxt = load_a16x32(arow, tkn * 32, half);
    v16bf b_nxt[4];
#pragma unroll
    for (int j = 0; j < 4; ++j)
      b_nxt[j] = *(const v16bf*)(Wbase + ((size_t)j * KT + tkn) * 512);
#pragma unroll
    for (int j = 0; j < 4; ++j)
      acc[j] = wmma_bf(a_cur, b_cur[j], acc[j]);
    a_cur = a_nxt;
#pragma unroll
    for (int j = 0; j < 4; ++j) b_cur[j] = b_nxt[j];
  }
}

// ---------------------------------------------------------------------------
// Kernel 3: QKV projections. One wave computes a 16(M) x 64(N) strip so the
// A operand is reused across 4 WMMAs per k-step. z = which weight.
// w in {0:Qs, 1:Ks, 2:Vs(transposed), 3:Ql, 4:Kl, 5:Vl(transposed)}.
// V stored as (B, H, HD, S) so attention's PV B-operand is contiguous.
// ---------------------------------------------------------------------------
__global__ __launch_bounds__(32)
void qkv_gemm(const bf16_t* __restrict__ xh,
              const bf16_t* __restrict__ Wp6,
              bf16_t* __restrict__ qkv) {
  const int lane = threadIdx.x;
  const int half = lane >> 4;
  const int ln   = lane & 15;
  const int tn4  = blockIdx.x;          // N strip of 64 (16 strips)
  const int m0   = blockIdx.y * 16;     // M rows (256 tiles)
  const int w    = blockIdx.z;          // weight (6)
  const int KT   = Dn / 32;             // 32

  const bf16_t* __restrict__ Wbase =
      Wp6 + (size_t)w * Dn * Dn + ((size_t)(tn4 * 4) * KT) * 512 + lane * 16;
  const bf16_t* __restrict__ arow = xh + (size_t)(m0 + ln) * Dn;

  v8f acc[4];
#pragma unroll
  for (int j = 0; j < 4; ++j) acc[j] = vzero8();

  gemm_strip_16x64(arow, Wbase, KT, half, acc);

  if (w == 2 || w == 5) {               // V: store transposed (B,H,HD,S)
    bf16_t* __restrict__ Vt = qkv + (size_t)w * SZE;
#pragma unroll
    for (int j = 0; j < 4; ++j) {
      int n = (tn4 * 4 + j) * 16 + ln, h = n >> 6, d = n & 63;
#pragma unroll
      for (int r = 0; r < 8; ++r) {
        int m = m0 + half * 8 + r;
        int bb = m >> 11;               // m / Sn
        int s  = m & (Sn - 1);
        Vt[(((size_t)bb * Hn + h) * HDn + d) * Sn + s] = f2bf(acc[j][r]);
      }
    }
  } else {                              // Q/K: natural (B*S, D)
    bf16_t* __restrict__ O = qkv + (size_t)w * SZE;
#pragma unroll
    for (int j = 0; j < 4; ++j) {
      int n0 = (tn4 * 4 + j) * 16;
#pragma unroll
      for (int r = 0; r < 8; ++r)
        O[(size_t)(m0 + half * 8 + r) * Dn + n0 + ln] = f2bf(acc[j][r]);
    }
  }
}

// ---------------------------------------------------------------------------
// Kernel 4: flash attention, 1 wave per (b, h, 16-query tile).
// SHORT=true : windowed causal + recency bias (only ~3 key blocks / tile)
// SHORT=false: full non-causal attention over all S keys (matches reference)
// Writes bf16 into combined (B*S, 2D): short -> cols [0,D), long -> [D,2D).
// ---------------------------------------------------------------------------
template <bool SHORT>
__global__ __launch_bounds__(32)
void attn_kernel(const bf16_t* __restrict__ Q,
                 const bf16_t* __restrict__ K,
                 const bf16_t* __restrict__ Vt,
                 bf16_t* __restrict__ comb) {
  __shared__ __align__(16) bf16_t Plds[16 * 32];

  const int lane = threadIdx.x;
  const int half = lane >> 4;
  const int ln   = lane & 15;
  const int q0   = blockIdx.x * 16;
  const int h    = blockIdx.y;
  const int b    = blockIdx.z;

  const bf16_t* __restrict__ Qb = Q + ((size_t)b * Sn) * Dn + (size_t)h * HDn;
  const bf16_t* __restrict__ Kb = K + ((size_t)b * Sn) * Dn + (size_t)h * HDn;
  const bf16_t* __restrict__ Vb = Vt + (((size_t)b * Hn + h) * HDn) * Sn;

  // Q tile A-operands for the two 32-wide contraction chunks of HD=64
  const bf16_t* qrow = Qb + (size_t)(q0 + ln) * Dn;
  v16bf aQ0 = load_a16x32(qrow, 0, half);
  v16bf aQ1 = load_a16x32(qrow, 32, half);

  float mrow[8], lrow[8];
  v8f acc[4];
#pragma unroll
  for (int r = 0; r < 8; ++r) { mrow[r] = NEGB; lrow[r] = 0.0f; }
#pragma unroll
  for (int t = 0; t < 4; ++t) acc[t] = vzero8();

  int kb_lo = 0, kb_hi = Sn / 32 - 1;
  if (SHORT) {
    int lo = (q0 - 64) >> 5;            // arithmetic shift = floor div
    kb_lo = lo > 0 ? lo : 0;
    kb_hi = (q0 + 15) >> 5;
  }

  for (int kb = kb_lo; kb <= kb_hi; ++kb) {
    const int k0 = kb * 32;

    // ---- S = Q K^T  (16 x 32 scores, two 16x16 WMMA column tiles) ----
    v8f s[2];
#pragma unroll
    for (int kt = 0; kt < 2; ++kt) {
      const bf16_t* krow = Kb + (size_t)(k0 + kt * 16 + ln) * Dn + half * 16;
      v16bf b0 = *(const v16bf*)(krow);
      v16bf b1 = *(const v16bf*)(krow + 32);
      v8f c = vzero8();
      c = wmma_bf(aQ0, b0, c);
      c = wmma_bf(aQ1, b1, c);
      s[kt] = c;
    }

    // ---- scale, recency bias, window mask ----
#pragma unroll
    for (int kt = 0; kt < 2; ++kt) {
#pragma unroll
      for (int r = 0; r < 8; ++r) {
        float v = s[kt][r] * SCALEF;
        if (SHORT) {
          int q   = q0 + half * 8 + r;
          int kk  = k0 + kt * 16 + ln;
          int rel = kk - q;
          v = (rel <= 0 && rel >= -64) ? (v + (float)rel * LOGD) : NEGB;
        }
        s[kt][r] = v;
      }
    }

    // ---- online softmax (row stats via shfl across the 16-lane half) ----
#pragma unroll
    for (int r = 0; r < 8; ++r) {
      float tmax = fmaxf(s[0][r], s[1][r]);
#pragma unroll
      for (int m = 1; m < 16; m <<= 1) tmax = fmaxf(tmax, __shfl_xor(tmax, m, 32));
      float mnew = fmaxf(mrow[r], tmax);
      float fac  = __expf(mrow[r] - mnew);
      float p0   = __expf(s[0][r] - mnew);
      float p1   = __expf(s[1][r] - mnew);
      float psum = p0 + p1;
#pragma unroll
      for (int m = 1; m < 16; m <<= 1) psum += __shfl_xor(psum, m, 32);
      lrow[r] = lrow[r] * fac + psum;
      mrow[r] = mnew;
#pragma unroll
      for (int t = 0; t < 4; ++t) acc[t][r] *= fac;
      s[0][r] = p0;
      s[1][r] = p1;
    }

    // ---- P (16x32) through LDS to re-layout as A-operand ----
#pragma unroll
    for (int kt = 0; kt < 2; ++kt)
#pragma unroll
      for (int r = 0; r < 8; ++r)
        Plds[(half * 8 + r) * 32 + kt * 16 + ln] = f2bf(s[kt][r]);
    asm volatile("s_wait_dscnt 0" ::: "memory");
    v8bf plo = *(const v8bf*)(Plds + ln * 32 + half * 16);
    v8bf phi = *(const v8bf*)(Plds + ln * 32 + half * 16 + 8);
    v16bf pa;
#pragma unroll
    for (int i = 0; i < 8; ++i) { pa[i] = plo[i]; pa[i + 8] = phi[i]; }

    // ---- O += P V  (V^T gives contiguous B-operand per dim tile) ----
#pragma unroll
    for (int t = 0; t < 4; ++t) {
      const bf16_t* vrow = Vb + (size_t)(t * 16 + ln) * Sn + k0 + half * 16;
      v16bf bv = *(const v16bf*)vrow;
      acc[t] = wmma_bf(pa, bv, acc[t]);
    }
    asm volatile("s_wait_dscnt 0" ::: "memory");  // P loads done before reuse
  }

  // ---- normalize and store into combined (B*S, 2D) ----
  float inv[8];
#pragma unroll
  for (int r = 0; r < 8; ++r) inv[r] = 1.0f / lrow[r];
  const int colbase = (SHORT ? 0 : Dn) + h * HDn;
#pragma unroll
  for (int t = 0; t < 4; ++t)
#pragma unroll
    for (int r = 0; r < 8; ++r) {
      int s = q0 + half * 8 + r;
      comb[((size_t)(b * Sn + s)) * (2 * Dn) + colbase + t * 16 + ln] =
          f2bf(acc[t][r] * inv[r]);
    }
}

// ---------------------------------------------------------------------------
// Kernel 5: output projection: (B*S, 2D) bf16 @ packed Wo (2D, D) + bo -> fp32
// One wave computes a 16(M) x 64(N) strip (A reused across 4 WMMAs / k-step).
// ---------------------------------------------------------------------------
__global__ __launch_bounds__(32)
void out_gemm(const bf16_t* __restrict__ comb,
              const bf16_t* __restrict__ Wop,
              const float* __restrict__ bo,
              float* __restrict__ out) {
  const int lane = threadIdx.x;
  const int half = lane >> 4;
  const int ln   = lane & 15;
  const int tn4  = blockIdx.x;          // N strip of 64 (16 strips)
  const int m0   = blockIdx.y * 16;
  const int KT   = (2 * Dn) / 32;       // 64

  const bf16_t* __restrict__ arow = comb + (size_t)(m0 + ln) * (2 * Dn);
  const bf16_t* __restrict__ Wbase =
      Wop + ((size_t)(tn4 * 4) * KT) * 512 + lane * 16;

  v8f acc[4];
#pragma unroll
  for (int j = 0; j < 4; ++j) acc[j] = vzero8();

  gemm_strip_16x64(arow, Wbase, KT, half, acc);

#pragma unroll
  for (int j = 0; j < 4; ++j) {
    int n = (tn4 * 4 + j) * 16 + ln;
    float bias = bo[n];
#pragma unroll
    for (int r = 0; r < 8; ++r)
      out[(size_t)(m0 + half * 8 + r) * Dn + n] = acc[j][r] + bias;
  }
}

// ---------------------------------------------------------------------------
// Host launch
// ---------------------------------------------------------------------------
extern "C" void kernel_launch(void* const* d_in, const int* in_sizes, int n_in,
                              void* d_out, int out_size, void* d_ws, size_t ws_size,
                              hipStream_t stream) {
  (void)in_sizes; (void)n_in; (void)out_size; (void)ws_size;

  const float* x  = (const float*)d_in[0];
  // d_in[1] = mask (all true) -> unused
  const float* Ws[6] = {(const float*)d_in[2], (const float*)d_in[3],
                        (const float*)d_in[4], (const float*)d_in[5],
                        (const float*)d_in[6], (const float*)d_in[7]};
  const float* Wo = (const float*)d_in[8];
  const float* bo = (const float*)d_in[9];
  float* out = (float*)d_out;

  char* ws = (char*)d_ws;
  bf16_t* xh   = (bf16_t*)(ws);                         //  8.0 MB
  bf16_t* Wp6  = (bf16_t*)(ws + 8388608);               // 12.0 MB
  bf16_t* Wop  = (bf16_t*)(ws + 20971520);              //  4.0 MB
  bf16_t* qkv  = (bf16_t*)(ws + 25165824);              // 48.0 MB
  bf16_t* comb = (bf16_t*)(ws + 75497472);              // 16.0 MB

  // 1) convert x to bf16
  {
    int n = Mrows * Dn;
    cvt_f32_bf16<<<(n + 255) / 256, 256, 0, stream>>>(x, xh, n);
  }
  // 2) pack weights into B-operand layout
  for (int w = 0; w < 6; ++w) {
    int n = (Dn / 16) * (Dn / 32) * 512;                // 1,048,576
    pack_w_b<<<(n + 255) / 256, 256, 0, stream>>>(Ws[w],
        Wp6 + (size_t)w * Dn * Dn, Dn);
  }
  {
    int n = (Dn / 16) * ((2 * Dn) / 32) * 512;          // 2,097,152
    pack_w_b<<<(n + 255) / 256, 256, 0, stream>>>(Wo, Wop, 2 * Dn);
  }
  // 3) six projections (Q/K natural, V transposed), 16x64 strips per wave
  qkv_gemm<<<dim3(Dn / 64, Mrows / 16, 6), 32, 0, stream>>>(xh, Wp6, qkv);
  // 4) attention
  attn_kernel<true ><<<dim3(Sn / 16, Hn, Bn), 32, 0, stream>>>(
      qkv + 0 * SZE, qkv + 1 * SZE, qkv + 2 * SZE, comb);
  attn_kernel<false><<<dim3(Sn / 16, Hn, Bn), 32, 0, stream>>>(
      qkv + 3 * SZE, qkv + 4 * SZE, qkv + 5 * SZE, comb);
  // 5) output projection + bias -> fp32, 16x64 strips per wave
  out_gemm<<<dim3(Dn / 64, Mrows / 16), 32, 0, stream>>>(comb, Wop, bo, out);
}
